// complexLSTM_69114613727355
// MI455X (gfx1250) — matrix operand order repaired and checked by reading, not compiled, not run on hardware
//
#include <hip/hip_runtime.h>

#define BATCH 512
#define EMBED 512
#define NENT  200000

#define BM 128
#define BN 64
#define KC 32
#define NKCH (EMBED / KC)     // 16 k-chunks
#define LDST 40               // LDS row stride in u16 (KC + 8 pad): 80B rows, 16B-aligned, conflict-free

typedef __attribute__((ext_vector_type(8)))  __bf16 v8bf;
typedef __attribute__((ext_vector_type(16))) __bf16 v16bf;
typedef __attribute__((ext_vector_type(8)))  float  v8f;

#define A_MAT (BM * LDST)           // 5120 u16 per A sub-matrix slice
#define A_BUF (4 * A_MAT)           // {A1h, A1l, A2h, A2l}
#define B_MAT (BN * LDST)           // 2560 u16 per B sub-matrix slice
#define B_BUF (4 * B_MAT)           // {Erh, Erl, Eih, Eil}
#define SMEM_U16 (2 * (A_BUF + B_BUF))
#define SMEM_BYTES (SMEM_U16 * 2)   // 122880 bytes (double buffered), < 320KB WGP LDS

#define WS_MAT (BATCH * EMBED)      // u16 elems per split matrix in workspace (ws uses 2 MB)

// Split: hi = bf16-truncate(x), lo = bf16(x - hi). x ~= hi + lo to ~16 mantissa bits.
__device__ __forceinline__ void split_f32(float x, unsigned short &h, unsigned short &l) {
  unsigned u = __float_as_uint(x);
  h = (unsigned short)(u >> 16);
  float hf = __uint_as_float(u & 0xFFFF0000u);
  float r  = x - hf;
  l = (unsigned short)(__float_as_uint(r) >> 16);
}

// Prologue: tmp1 = e1_r*r_r - e1_i*r_i ; tmp2 = e1_i*r_r + e1_r*r_i ; split to bf16 hi/lo in ws.
__global__ __launch_bounds__(256) void complex_prep_kernel(
    const float* __restrict__ e1r, const float* __restrict__ e1i,
    const float* __restrict__ rr,  const float* __restrict__ ri,
    unsigned short* __restrict__ ws) {
  int i = blockIdx.x * 256 + threadIdx.x;
  float ar = e1r[i], ai = e1i[i], br = rr[i], bi = ri[i];
  float t1 = ar * br - ai * bi;
  float t2 = ai * br + ar * bi;
  unsigned short h1, l1, h2, l2;
  split_f32(t1, h1, l1);
  split_f32(t2, h2, l2);
  ws[0 * WS_MAT + i] = h1;
  ws[1 * WS_MAT + i] = l1;
  ws[2 * WS_MAT + i] = h2;
  ws[3 * WS_MAT + i] = l2;
}

__device__ __forceinline__ v16bf frag16(const unsigned short* p0, const unsigned short* p1) {
  v8bf x = *(const v8bf*)p0;   // ds_read_b128
  v8bf y = *(const v8bf*)p1;   // ds_read_b128
  return __builtin_shufflevector(x, y, 0,1,2,3,4,5,6,7,8,9,10,11,12,13,14,15);
}

#define WMMA_BF16(Af, Bf_, Cacc) \
  __builtin_amdgcn_wmma_f32_16x16x32_bf16(false, (Af), false, (Bf_), (short)0, (Cacc), false, false)

__global__ __launch_bounds__(256) void complex_score_kernel(
    const unsigned short* __restrict__ ws,
    const float* __restrict__ e2r,
    const float* __restrict__ e2i,
    float* __restrict__ out) {
  extern __shared__ unsigned short smem[];
  unsigned short* sA = smem;              // [2 buf][4 mat][BM][LDST]
  unsigned short* sB = smem + 2 * A_BUF;  // [2 buf][4 mat][BN][LDST]

  const int t    = threadIdx.x;
  const int lane = t & 31;                // wave32
  const int wave = t >> 5;                // 8 waves
  const int wm   = wave & 3;              // 4 M-groups of 32 rows
  const int wn   = wave >> 2;             // 2 N-groups of 32 cols
  const int lrow = lane & 15;
  const int m0   = blockIdx.x * BM;
  const int n0   = blockIdx.y * BN;

  const int a_c = t & 3, a_r = t >> 2;    // A staging: 16B chunk c, row r (+64 for pass 1)
  const int b_c = t & 7, b_r = t >> 3;    // B staging: float4 chunk c, row r (+32 for pass 1)

  float4 bReg[2][2];                      // only B stays staged across compute (HBM latency)
  const float* eMat[2] = { e2r, e2i };

  // A workspace is 2MB and L2-resident: load + commit to LDS immediately (short reg lifetime).
  auto load_commit_A = [&](int k0, int buf) {
    unsigned short* dA = sA + buf * A_BUF;
#pragma unroll
    for (int m = 0; m < 4; ++m) {
      const unsigned short* p = ws + m * WS_MAT + (m0 + a_r) * EMBED + k0 + a_c * 8;
      uint4 x0 = *(const uint4*)p;
      uint4 x1 = *(const uint4*)(p + 64 * EMBED);
      unsigned short* q = dA + m * A_MAT + a_r * LDST + a_c * 8;
      *(uint4*)q = x0;
      *(uint4*)(q + 64 * LDST) = x1;
    }
  };

  auto stage_B = [&](int k0) {
#pragma unroll
    for (int m = 0; m < 2; ++m) {
      const float* p = eMat[m] + (size_t)(n0 + b_r) * EMBED + k0 + b_c * 4;
      bReg[m][0] = *(const float4*)p;
      bReg[m][1] = *(const float4*)(p + (size_t)32 * EMBED);
    }
  };

  auto commit_B = [&](int buf) {
    unsigned short* dB = sB + buf * B_BUF;
#pragma unroll
    for (int m = 0; m < 2; ++m) {
#pragma unroll
      for (int p = 0; p < 2; ++p) {
        float4 v = bReg[m][p];
        ushort4 h, l;
        split_f32(v.x, h.x, l.x);
        split_f32(v.y, h.y, l.y);
        split_f32(v.z, h.z, l.z);
        split_f32(v.w, h.w, l.w);
        unsigned short* qh = dB + (2 * m) * B_MAT + (b_r + p * 32) * LDST + b_c * 4;
        *(ushort4*)qh = h;
        *(ushort4*)(qh + B_MAT) = l;
      }
    }
  };

  v8f acc[2][2];
  {
    v8f z = {0.f, 0.f, 0.f, 0.f, 0.f, 0.f, 0.f, 0.f};
#pragma unroll
    for (int i = 0; i < 2; ++i)
#pragma unroll
      for (int j = 0; j < 2; ++j) acc[i][j] = z;
  }

  // Fragment addressing used here, contiguous k, same for A and B (16-bit A 16x32 / B 32x16, wave32):
  // A: lanes 0-15 hold K{0..15},       lanes 16-31 hold K{16..31}
  // B: lanes 0-15 hold K{0..15},       lanes 16-31 hold K{16..31}
  const int aoff = (lane < 16) ? 0 : 16;
  const int boff = (lane < 16) ? 0 : 16;

  auto do_compute = [&](int cur) {
    const unsigned short* cA = sA + cur * A_BUF;
    const unsigned short* cB = sB + cur * B_BUF;
    // Hoist all A fragments (reused by both nt); B fragments loaded per-nt to cap live VGPRs.
    v16bf Af[2][4];   // [mt][{A1h, A1l, A2h, A2l}]
#pragma unroll
    for (int mt = 0; mt < 2; ++mt) {
      const int ar_ = (wm * 32 + mt * 16 + lrow) * LDST;
#pragma unroll
      for (int m = 0; m < 4; ++m)
        Af[mt][m] = frag16(cA + m * A_MAT + ar_ + aoff,
                           cA + m * A_MAT + ar_ + aoff + 8);
    }
#pragma unroll
    for (int nt = 0; nt < 2; ++nt) {
      const int br_ = (wn * 32 + nt * 16 + lrow) * LDST;
      v16bf brh = frag16(cB + 0 * B_MAT + br_ + boff, cB + 0 * B_MAT + br_ + boff + 8);
      v16bf brl = frag16(cB + 1 * B_MAT + br_ + boff, cB + 1 * B_MAT + br_ + boff + 8);
      v16bf bih = frag16(cB + 2 * B_MAT + br_ + boff, cB + 2 * B_MAT + br_ + boff + 8);
      v16bf bil = frag16(cB + 3 * B_MAT + br_ + boff, cB + 3 * B_MAT + br_ + boff + 8);
#pragma unroll
      for (int mt = 0; mt < 2; ++mt) {
        v8f c = acc[mt][nt];
        c = WMMA_BF16(Af[mt][0], brh, c);   // real: Ah*Bh
        c = WMMA_BF16(Af[mt][1], brh, c);   //       Al*Bh
        c = WMMA_BF16(Af[mt][0], brl, c);   //       Ah*Bl
        c = WMMA_BF16(Af[mt][2], bih, c);   // imag: Ah*Bh
        c = WMMA_BF16(Af[mt][3], bih, c);   //       Al*Bh
        c = WMMA_BF16(Af[mt][2], bil, c);   //       Ah*Bl
        acc[mt][nt] = c;
      }
    }
  };

  // Prologue: fill buffer 0.
  stage_B(0);
  load_commit_A(0, 0);
  commit_B(0);
  __syncthreads();

  for (int kt = 0; kt < NKCH; ++kt) {
    const int cur = kt & 1;
    if (kt + 1 < NKCH) {
      stage_B((kt + 1) * KC);               // HBM-latency loads in flight during compute
      load_commit_A((kt + 1) * KC, cur ^ 1);// L2-resident: commit now, regs die before compute
    }
    if (kt + 2 < NKCH) {                    // warm the entity stream (global_prefetch_b8)
      const int k2 = (kt + 2) * KC;
      __builtin_prefetch(e2r + (size_t)(n0 + b_r) * EMBED + k2 + b_c * 4, 0, 1);
      __builtin_prefetch(e2i + (size_t)(n0 + b_r) * EMBED + k2 + b_c * 4, 0, 1);
    }
    do_compute(cur);
    if (kt + 1 < NKCH) commit_B(cur ^ 1);
    __syncthreads();
  }

  // C/D layout: 8 VGPRs; lanes 0-15 -> M = v, lanes 16-31 -> M = v + 8; N = lane % 16
  const int rbase = m0 + wm * 32 + ((lane < 16) ? 0 : 8);
  const int cbase = n0 + wn * 32 + lrow;
#pragma unroll
  for (int mt = 0; mt < 2; ++mt)
#pragma unroll
    for (int nt = 0; nt < 2; ++nt) {
      v8f c = acc[mt][nt];
#pragma unroll
      for (int v = 0; v < 8; ++v) {
        int row = rbase + mt * 16 + v;
        int col = cbase + nt * 16;
        out[(size_t)row * NENT + col] = c[v];
      }
    }
}

extern "C" void kernel_launch(void* const* d_in, const int* in_sizes, int n_in,
                              void* d_out, int out_size, void* d_ws, size_t ws_size,
                              hipStream_t stream) {
  const float* e1r = (const float*)d_in[0];
  const float* e1i = (const float*)d_in[1];
  const float* rr  = (const float*)d_in[2];
  const float* ri  = (const float*)d_in[3];
  const float* e2r = (const float*)d_in[4];
  const float* e2i = (const float*)d_in[5];
  float* out = (float*)d_out;
  unsigned short* ws = (unsigned short*)d_ws;   // 4 split bf16 matrices, 2 MB

  complex_prep_kernel<<<dim3((BATCH * EMBED) / 256), 256, 0, stream>>>(e1r, e1i, rr, ri, ws);

  // x = B-blocks fastest so the 4 row-blocks sharing an entity tile are co-resident (L2 reuse)
  dim3 grid(BATCH / BM, NENT / BN);   // (4, 3125), exact tiling
  complex_score_kernel<<<grid, 256, SMEM_BYTES, stream>>>(ws, e2r, e2i, out);
}
